// CriticNetwork_1_26972394619302
// MI455X (gfx1250) — matrix operand order repaired and checked by reading, not compiled, not run on hardware
//
#include <hip/hip_runtime.h>
#include <math.h>

typedef _Float16 f16;
typedef __attribute__((ext_vector_type(16))) _Float16 v16h;
typedef __attribute__((ext_vector_type(8)))  _Float16 v8h;
typedef __attribute__((ext_vector_type(8)))  float    v8f;

#define NB 512
#define NN 1024
#define HH 128

// LDS layout (bytes):
//   u2    : NN*HH*2   = 262144   (f16, attention values, loop-invariant)
//   wt    : HH*HH*2   =  32768   (f16, W_ref transposed: wt[n*HH+k])
//   uvals : NN*4      =   4096   (f32, logits / exp weights)
//   sWemb : 2*HH*4    =   1024
//   sbemb : HH*4      =    512
//   sbref : HH*4      =    512
//   sVec  : HH*4      =    512
//   query : HH*4      =    512
//   u1    : HH*4      =    512
//   red   : 256*4     =   1024
#define SMEM_BYTES (NN*HH*2 + HH*HH*2 + NN*4 + 2*HH*4 + 5*HH*4 + 256*4)

__device__ __forceinline__ float fast_tanh(float x) {
#if __has_builtin(__builtin_amdgcn_tanhf)
    return __builtin_amdgcn_tanhf(x);       // native v_tanh_f32 (CDNA5 TRANS op)
#elif __has_builtin(__builtin_amdgcn_tanh_f32)
    return __builtin_amdgcn_tanh_f32(x);
#else
    const float e = __expf(2.0f * x);       // v_exp_f32 path
    return (e - 1.0f) / (e + 1.0f);
#endif
}

__global__ __launch_bounds__(256) void critic_fused_kernel(
    const float* __restrict__ x,      // (B,N,2)
    const float* __restrict__ W_emb,  // (2,H)
    const float* __restrict__ b_emb,  // (H)
    const float* __restrict__ dec_in, // (H)
    const float* __restrict__ Vec,    // (H)
    const float* __restrict__ W_q,    // (H,H)
    const float* __restrict__ b_q,    // (H)
    const float* __restrict__ W_ref,  // (H,H)
    const float* __restrict__ b_ref,  // (H)
    const float* __restrict__ W_fc1,  // (H,H)
    const float* __restrict__ W_fc2,  // (H,1)
    float* __restrict__ out)          // (B)
{
    extern __shared__ char smem[];
    f16*   u2    = (f16*)smem;                              // NN*HH
    f16*   wt    = (f16*)(smem + NN*HH*2);                  // HH*HH
    float* uvals = (float*)(smem + NN*HH*2 + HH*HH*2);      // NN
    float* sWemb = uvals + NN;                              // 2*HH
    float* sbemb = sWemb + 2*HH;                            // HH
    float* sbref = sbemb + HH;                              // HH
    float* sVec  = sbref + HH;                              // HH
    float* query = sVec  + HH;                              // HH
    float* u1    = query + HH;                              // HH
    float* red   = u1    + HH;                              // 256

    const int tid  = threadIdx.x;
    const int b    = blockIdx.x;
    const int lane = tid & 31;
    const int wave = tid >> 5;
    const int hcol = tid & (HH - 1);   // 0..127, output column for split GEMVs
    const int half = tid >> 7;         // 0/1, reduction-dim half for split GEMVs

    // ---- Stage weights into LDS ----
    for (int i = tid; i < HH*HH; i += 256) {
        int k = i >> 7, n = i & (HH-1);        // coalesced read of W_ref[k][n]
        wt[n*HH + k] = (f16)W_ref[i];          // store transposed
    }
    for (int i = tid; i < 2*HH; i += 256) sWemb[i] = W_emb[i];
    if (tid < HH) {
        sbemb[tid] = b_emb[tid];
        sbref[tid] = b_ref[tid];
        sVec[tid]  = Vec[tid];
        query[tid] = dec_in[tid];
    }
    __syncthreads();

    // ---- Phase 1: u2 = (x@W_emb + b_emb) @ W_ref + b_ref  (WMMA, f16 -> LDS)
    // A frag (16x32 f16): lane m=lane&15, h=lane>>4; elem e -> K=h*16+e.
    // B frag (32x16): lane n=lane&15, h=lane>>4;
    // elem e -> K=h*16+e (contiguous run in transposed W_ref).
    const int ml = lane & 15;
    const int hf = lane >> 4;
    for (int mt = wave * 8; mt < wave * 8 + 8; ++mt) {
        const int row = mt * 16 + ml;
        const float x0 = x[(size_t)b * NN * 2 + row * 2 + 0];
        const float x1 = x[(size_t)b * NN * 2 + row * 2 + 1];

        v8f acc[8];
        #pragma unroll
        for (int nt = 0; nt < 8; ++nt) acc[nt] = (v8f){0,0,0,0,0,0,0,0};

        #pragma unroll
        for (int ks = 0; ks < 4; ++ks) {
            v16h afrag;
            #pragma unroll
            for (int e = 0; e < 8; ++e) {
                int k  = ks * 32 + hf * 16 + e;
                int k2 = k + 8;
                afrag[e]     = (f16)(x0 * sWemb[k]  + x1 * sWemb[HH + k]  + sbemb[k]);
                afrag[e + 8] = (f16)(x0 * sWemb[k2] + x1 * sWemb[HH + k2] + sbemb[k2]);
            }
            #pragma unroll
            for (int nt = 0; nt < 8; ++nt) {
                v16h bfrag;
                const f16* wp = &wt[(nt * 16 + ml) * HH + ks * 32 + hf * 16];
                #pragma unroll
                for (int e = 0; e < 16; ++e) bfrag[e] = wp[e];
                acc[nt] = __builtin_amdgcn_wmma_f32_16x16x32_f16(
                    false, afrag, false, bfrag, (short)0, acc[nt], false, false);
            }
        }
        // C/D layout: VGPR r -> M = r + 8*hf (tile-local), N = lane&15
        #pragma unroll
        for (int nt = 0; nt < 8; ++nt) {
            const int cc = nt * 16 + ml;
            const float br = sbref[cc];
            #pragma unroll
            for (int r = 0; r < 8; ++r) {
                const int rr = mt * 16 + r + hf * 8;
                u2[rr * HH + cc] = (f16)(acc[nt][r] + br);
            }
        }
    }
    __syncthreads();

    // ---- Phase 2: two glimpse iterations, u2 in LDS, W_q/b_q read from global ----
    for (int g = 0; g < 2; ++g) {
        // u1 = query @ W_q + b_q : split K over the two thread-halves
        {
            float s = 0.f;
            const int k0 = half * (HH / 2);
            for (int k = k0; k < k0 + HH / 2; ++k)
                s += query[k] * W_q[k * HH + hcol];
            red[tid] = s;
        }
        __syncthreads();
        if (tid < HH) u1[tid] = red[tid] + red[tid + HH] + b_q[tid];
        __syncthreads();

        // logits u[n] = dot(tanh(u1 + u2[n]), Vec) ; vectorized LDS row reads
        float uv[NN / 256];
        float lmax = -1e30f;
        #pragma unroll
        for (int j = 0; j < NN / 256; ++j) {
            const int n = tid + j * 256;
            const v8h* rowp = (const v8h*)&u2[n * HH];
            float s = 0.f;
            for (int c = 0; c < HH / 8; ++c) {
                v8h vv = rowp[c];
                #pragma unroll
                for (int e = 0; e < 8; ++e) {
                    const int k = c * 8 + e;
                    s += fast_tanh(u1[k] + (float)vv[e]) * sVec[k];
                }
            }
            uv[j] = s;
            lmax = fmaxf(lmax, s);
        }
        // block max
        red[tid] = lmax; __syncthreads();
        for (int s = 128; s > 0; s >>= 1) {
            if (tid < s) red[tid] = fmaxf(red[tid], red[tid + s]);
            __syncthreads();
        }
        const float mx = red[0];
        __syncthreads();
        // exp + block sum
        float lsum = 0.f;
        #pragma unroll
        for (int j = 0; j < NN / 256; ++j) {
            const int n = tid + j * 256;
            const float e = __expf(uv[j] - mx);
            uvals[n] = e;
            lsum += e;
        }
        red[tid] = lsum; __syncthreads();
        for (int s = 128; s > 0; s >>= 1) {
            if (tid < s) red[tid] += red[tid + s];
            __syncthreads();
        }
        const float inv = 1.f / red[0];
        __syncthreads();

        // query[h] = inv * sum_n exp_u[n] * u2[n][h] : split N over thread-halves
        {
            float s = 0.f;
            const int n0 = half * (NN / 2);
            for (int n = n0; n < n0 + NN / 2; ++n)
                s += uvals[n] * (float)u2[n * HH + hcol];
            red[tid] = s;
        }
        __syncthreads();
        if (tid < HH) query[tid] = (red[tid] + red[tid + HH]) * inv;
        __syncthreads();
    }

    // ---- Phase 3: pred = relu(query @ W_fc1) @ W_fc2 ----
    float rloc = 0.f;
    if (tid < HH) {
        float s = 0.f;
        for (int k = 0; k < HH; ++k) s += query[k] * W_fc1[k * HH + tid];
        rloc = fmaxf(s, 0.f) * W_fc2[tid];
    }
    red[tid] = rloc; __syncthreads();
    for (int s = 128; s > 0; s >>= 1) {
        if (tid < s) red[tid] += red[tid + s];
        __syncthreads();
    }
    if (tid == 0) out[b] = red[0];
}

extern "C" void kernel_launch(void* const* d_in, const int* in_sizes, int n_in,
                              void* d_out, int out_size, void* d_ws, size_t ws_size,
                              hipStream_t stream) {
    (void)in_sizes; (void)n_in; (void)out_size; (void)d_ws; (void)ws_size;
    // setup_inputs order:
    // 0:x 1:mask 2:W_emb 3:b_emb 4:dec_input 5:Vec 6:W_q 7:b_q 8:W_ref 9:b_ref 10:W_fc1 11:W_fc2
    const float* x      = (const float*)d_in[0];
    const float* W_emb  = (const float*)d_in[2];
    const float* b_emb  = (const float*)d_in[3];
    const float* dec_in = (const float*)d_in[4];
    const float* Vec    = (const float*)d_in[5];
    const float* W_q    = (const float*)d_in[6];
    const float* b_q    = (const float*)d_in[7];
    const float* W_ref  = (const float*)d_in[8];
    const float* b_ref  = (const float*)d_in[9];
    const float* W_fc1  = (const float*)d_in[10];
    const float* W_fc2  = (const float*)d_in[11];
    float* out = (float*)d_out;

    // Opt into large dynamic LDS (320 KB/WGP on CDNA5). Deterministic, capture-safe.
    (void)hipFuncSetAttribute((const void*)critic_fused_kernel,
                              hipFuncAttributeMaxDynamicSharedMemorySize, SMEM_BYTES);

    critic_fused_kernel<<<NB, 256, SMEM_BYTES, stream>>>(
        x, W_emb, b_emb, dec_in, Vec, W_q, b_q, W_ref, b_ref, W_fc1, W_fc2, out);
}
